// TransFusion2_83614423318580
// MI455X (gfx1250) — matrix operand order repaired and checked by reading, not compiled, not run on hardware
//
#include <hip/hip_runtime.h>

// ---------------------------------------------------------------------------
// CDNA5 (gfx1250) TransFusion2.
// Heavy GEMMs on v_wmma_f32_16x16x32_bf16 (bf16 in, f32 accum).
// Local windowed attention (no softmax) refactored as linear attention:
//   o = q (k^T v)  ->  o = (x2w+pos) @ (qw @ blockdiag(M_h)) + qb @ BD
// Fragment I/O is vectorized: A-fragments = 2x b128 loads (K-contiguous),
// B-fragments come from weights pre-swizzled into per-lane packed order.
// ---------------------------------------------------------------------------

#define DEV static __device__ __forceinline__

typedef __attribute__((ext_vector_type(16))) __bf16 v16bf;
typedef __attribute__((ext_vector_type(8)))  __bf16 v8bf;
typedef __attribute__((ext_vector_type(8)))  float  v8f;

DEV unsigned short f2bf(float f) {
  unsigned u = __builtin_bit_cast(unsigned, f);
  return (unsigned short)((u + 0x7FFFu + ((u >> 16) & 1u)) >> 16);
}
DEV __bf16 f2bfh(float f) { return __builtin_bit_cast(__bf16, f2bf(f)); }
DEV float bf2f(unsigned short u) {
  unsigned x = ((unsigned)u) << 16;
  return __builtin_bit_cast(float, x);
}

// ---- A fragment (16x32 bf16), contiguous k order used here (B matches):
//   lane L: m=L&15, kbase=(L<16)?0:16, element j -> k = kbase + j
// Generic (element-wise, supports transpose + K bounds):
DEV v16bf frag_a(const unsigned short* base, int strideM, int strideK,
                 int k0, int kLimit, int lane) {
  int m = lane & 15;
  int kbase = (lane < 16) ? 0 : 16;
  v16bf a;
#pragma unroll
  for (int j = 0; j < 16; ++j) {
    int k = k0 + kbase + j;
    unsigned short u = (k < kLimit) ? base[m * strideM + k * strideK] : (unsigned short)0;
    a[j] = __builtin_bit_cast(__bf16, u);
  }
  return a;
}
// Vectorized (strideK==1, fully in-bounds): two 16B loads per lane.
DEV v16bf frag_a_v(const unsigned short* base, int strideM, int k0, int lane) {
  int m = lane & 15;
  int kbase = (lane < 16) ? 0 : 16;
  const unsigned short* p = base + m * strideM + k0 + kbase;
  v8bf lo = *(const v8bf*)p;
  v8bf hi = *(const v8bf*)(p + 8);
  return __builtin_shufflevector(lo, hi, 0, 1, 2, 3, 4, 5, 6, 7,
                                 8, 9, 10, 11, 12, 13, 14, 15);
}

// ---- B fragment (32x16 bf16): lane L: n=L&15, element j -> k = (L<16)?j:16+j
// Generic (element-wise gather down a column):
DEV v16bf frag_b(const unsigned short* base, int strideK, int k0, int kLimit, int lane) {
  int n = lane & 15;
  v16bf b;
#pragma unroll
  for (int j = 0; j < 16; ++j) {
    int k = k0 + ((lane < 16) ? j : (16 + j));
    unsigned short u = (k < kLimit) ? base[k * strideK + n] : (unsigned short)0;
    b[j] = __builtin_bit_cast(__bf16, u);
  }
  return b;
}
// Packed layout [ct][kt][lane][16]: one lane's fragment = 32 contiguous bytes.
DEV v16bf frag_b_pk(const unsigned short* pk, int ct, int kt, int KT, int lane) {
  const unsigned short* p = pk + ((((ct * KT) + kt) * 32 + lane) << 4);
  v8bf lo = *(const v8bf*)p;
  v8bf hi = *(const v8bf*)(p + 8);
  return __builtin_shufflevector(lo, hi, 0, 1, 2, 3, 4, 5, 6, 7,
                                 8, 9, 10, 11, 12, 13, 14, 15);
}

DEV v8f wmma_step(v16bf a, v16bf b, v8f c) {
  return __builtin_amdgcn_wmma_f32_16x16x32_bf16(false, a, false, b, (short)0, c, false, false);
}

// ---------------------------------------------------------------------------
// Positional sine tables (posq 24x24x128, posk 12x12x256, posg 8x8x128)
// ---------------------------------------------------------------------------
DEV float pos_val(int hsz, int wsz, int csz, int yi, int xi, int ci) {
  int npf = csz >> 1;
  float v; int j;
  if (ci < npf) { v = (float)(yi + 1) / ((float)hsz + 1e-5f) * 6.283185307179586f; j = ci; }
  else          { v = (float)(xi + 1) / ((float)wsz + 1e-5f) * 6.283185307179586f; j = ci - npf; }
  float dim_t = powf(10000.f, (2.f * (float)(j >> 1)) / (float)npf);
  float p = v / dim_t;
  return (j & 1) ? cosf(p) : sinf(p);
}

__global__ void k_pos_init(float* posq, float* posk, float* posg) {
  int i = blockIdx.x * blockDim.x + threadIdx.x;
  if (i < 73728) {
    int yi = i / (24 * 128), xi = (i / 128) % 24, ci = i & 127;
    posq[i] = pos_val(24, 24, 128, yi, xi, ci);
  } else if (i < 73728 + 36864) {
    int k = i - 73728;
    int yi = k / (12 * 256), xi = (k / 256) % 12, ci = k & 255;
    posk[k] = pos_val(12, 12, 256, yi, xi, ci);
  } else if (i < 73728 + 36864 + 8192) {
    int k = i - 73728 - 36864;
    int yi = k / (8 * 128), xi = (k / 128) % 8, ci = k & 127;
    posg[k] = pos_val(8, 8, 128, yi, xi, ci);
  }
}

// plain f32 -> bf16 (row-major), used for A-side weights (qw)
__global__ void k_cvt_bf16(const float* __restrict__ src, unsigned short* __restrict__ dst, int n) {
  int i = blockIdx.x * blockDim.x + threadIdx.x;
  if (i < n) dst[i] = f2bf(src[i]);
}

// swizzle W (K x 128 f32) into B-fragment-packed bf16: [ct][kt][lane][16]
__global__ void k_pack_b(const float* __restrict__ W, unsigned short* __restrict__ pk, int K) {
  int i = blockIdx.x * blockDim.x + threadIdx.x;
  if (i >= K * 128) return;
  int KT = K >> 5;
  int j = i & 15;
  int lane = (i >> 4) & 31;
  int tmp = i >> 9;                 // ct*KT + kt
  int kt = tmp % KT, ct = tmp / KT;
  int k = kt * 32 + ((lane < 16) ? j : 16 + j);
  int n = ct * 16 + (lane & 15);
  pk[i] = f2bf(W[k * 128 + n]);
}

// ---------------------------------------------------------------------------
// shortcut = LN(x1 @ proj_w + proj_b)   rows = B*96*96 = 18432, K=256, N=128
// ---------------------------------------------------------------------------
__global__ __launch_bounds__(256) void k_proj_ln(
    const float* __restrict__ x1, const unsigned short* __restrict__ pw_pk,
    const float* __restrict__ pb, const float* __restrict__ gamma,
    const float* __restrict__ beta, float* __restrict__ shortcut) {
  __shared__ __align__(16) char smem[32768];
  unsigned short* Ast = (unsigned short*)smem;      // 32 x 256 bf16
  float* outS = (float*)(smem + 16384);             // 32 x 128 f32
  int tid = threadIdx.x, lane = tid & 31, wave = tid >> 5;
  int rows0 = blockIdx.x * 32;
  for (int idx = tid; idx < 32 * 256; idx += 256) {
    int r = idx & 31, ch = idx >> 5;
    int R = rows0 + r;
    int bi = R / 9216, rem = R % 9216;
    int hh = rem / 96, ww = rem % 96;
    Ast[r * 256 + ch] = f2bf(x1[((bi * 256 + ch) * 96 + hh) * 96 + ww]);
  }
  __syncthreads();
  for (int job = wave; job < 16; job += 8) {
    int rt = job >> 3, ct = job & 7;
    v8f acc = {};
    for (int k0 = 0; k0 < 256; k0 += 32)
      acc = wmma_step(frag_a_v(Ast + rt * 16 * 256, 256, k0, lane),
                      frag_b_pk(pw_pk, ct, k0 >> 5, 8, lane), acc);
    int n = lane & 15, mb = (lane < 16) ? 0 : 8;
#pragma unroll
    for (int j = 0; j < 8; ++j) {
      int row = rt * 16 + mb + j, col = ct * 16 + n;
      outS[row * 128 + col] = acc[j] + pb[col];
    }
  }
  __syncthreads();
  if (tid < 32) {
    float mu = 0.f;
    for (int c = 0; c < 128; ++c) mu += outS[tid * 128 + c];
    mu *= (1.f / 128.f);
    float var = 0.f;
    for (int c = 0; c < 128; ++c) { float d = outS[tid * 128 + c] - mu; var += d * d; }
    var *= (1.f / 128.f);
    float inv = rsqrtf(var + 1e-5f);
    int R = rows0 + tid;
    for (int c = 0; c < 128; ++c)
      shortcut[R * 128 + c] = (outS[tid * 128 + c] - mu) * inv * gamma[c] + beta[c];
  }
}

// ---------------------------------------------------------------------------
// Local windowed linear attention. One block per (batch, window) = 128 blocks.
// Phase A: k/v projections (WMMA) + M_h = k_h^T v_h (WMMA, one head per wave).
// Phase B: W' = qw @ blockdiag(M_h) (WMMA, K=16) stored B-fragment-packed.
// Phase C: qbuf = x2 + (x2w+pos) @ W' + b'  (WMMA, K=128).
// ---------------------------------------------------------------------------
__global__ __launch_bounds__(256) void k_local_attn(
    const float* __restrict__ x1, const float* __restrict__ x2,
    const unsigned short* __restrict__ kw_pk, const float* __restrict__ kbias,
    const unsigned short* __restrict__ vw_pk, const float* __restrict__ vbias,
    const unsigned short* __restrict__ qw_bf, const float* __restrict__ qbias,
    const float* __restrict__ posk, const float* __restrict__ posq,
    float* __restrict__ qbuf) {
  __shared__ __align__(16) char smem[41472];
  unsigned short* kvS = (unsigned short*)smem;            // phase A: 32x256 bf16
  unsigned short* kS  = (unsigned short*)(smem + 16384);  // phase A: 32x128 bf16
  unsigned short* vS  = (unsigned short*)(smem + 24576);  // phase A: 32x128 bf16
  unsigned short* Wp  = (unsigned short*)smem;            // phase B/C: 128x128 packed (aliases A)
  unsigned short* Ms  = (unsigned short*)(smem + 32768);  // 8 heads x 16x16 bf16
  float* bprime       = (float*)(smem + 36864);           // 128 f32
  unsigned short* Ast = (unsigned short*)(smem + 37376);  // 16x128 bf16

  int tid = threadIdx.x, lane = tid & 31, wave = tid >> 5;
  int blk = blockIdx.x;
  int bi = blk >> 6, win = blk & 63;
  int wy = win >> 3, wx = win & 7;

  // ---- Phase A ----
  v8f Macc = {};
  for (int c0 = 0; c0 < 160; c0 += 32) {          // 144 tokens in 5 chunks of 32
    for (int idx = tid; idx < 32 * 256; idx += 256) {
      int t = idx & 31, ch = idx >> 5;
      int tt = c0 + t;
      float v = 0.f;
      if (tt < 144) {
        int hh = wy * 12 + tt / 12, ww = wx * 12 + tt % 12;
        v = x1[((bi * 256 + ch) * 96 + hh) * 96 + ww] + posk[tt * 256 + ch];
        if (tt + 32 < 144) {
          int t2 = tt + 32;
          __builtin_prefetch(&x1[((bi * 256 + ch) * 96 + wy * 12 + t2 / 12) * 96 +
                                 wx * 12 + t2 % 12], 0, 1);
        }
      }
      kvS[t * 256 + ch] = f2bf(v);
    }
    __syncthreads();
    // k/v projection tiles: 32 jobs (mat, rowtile, coltile), K=256
    for (int job = wave; job < 32; job += 8) {
      int mat = job >> 4, rt = (job >> 3) & 1, ct = job & 7;
      const unsigned short* Wg = mat ? vw_pk : kw_pk;
      const float* bias = mat ? vbias : kbias;
      v8f acc = {};
      for (int k0 = 0; k0 < 256; k0 += 32)
        acc = wmma_step(frag_a_v(kvS + rt * 16 * 256, 256, k0, lane),
                        frag_b_pk(Wg, ct, k0 >> 5, 8, lane), acc);
      unsigned short* dst = mat ? vS : kS;
      int n = lane & 15, mb = (lane < 16) ? 0 : 8;
#pragma unroll
      for (int j = 0; j < 8; ++j) {
        int row = rt * 16 + mb + j, col = ct * 16 + n;
        float outv = ((c0 + row) < 144) ? (acc[j] + bias[col]) : 0.f;  // zero pad tokens
        dst[row * 128 + col] = f2bf(outv);
      }
    }
    __syncthreads();
    // M_h += k_h^T (16x32) @ v_h (32x16); transposed-A via stride swap
    Macc = wmma_step(frag_a(kS + wave * 16, 1, 128, 0, 32, lane),
                     frag_b(vS + wave * 16, 128, 0, 32, lane), Macc);
    __syncthreads();
  }

  // ---- Phase B ----
  {
    int n = lane & 15, mb = (lane < 16) ? 0 : 8;
#pragma unroll
    for (int j = 0; j < 8; ++j)
      Ms[wave * 256 + (mb + j) * 16 + n] = f2bf(Macc[j]);
  }
  __syncthreads();
  // W'[rt, head] tiles: 64 jobs, K=16 (zero-padded to 32 by the frag loaders).
  // Epilogue stores straight into B-fragment-packed layout (KT=4):
  //   consumer index for (k,n): ct=n>>4, kt=k>>5, laneB=((k>>4)&1)*16 + (n&15), jB=k&15
  for (int job = wave; job < 64; job += 8) {
    int rt = job >> 3, h = job & 7;
    v8f acc = {};
    acc = wmma_step(frag_a(qw_bf + (rt * 16) * 128 + h * 16, 128, 1, 0, 16, lane),
                    frag_b(Ms + h * 256, 16, 0, 16, lane), acc);
    int nl = lane & 15, mb = (lane < 16) ? 0 : 8;
    int bidx = ((h * 4 + (rt >> 1)) * 32 + (rt & 1) * 16 + nl) * 16 + mb;
    v8bf pkv;
#pragma unroll
    for (int j = 0; j < 8; ++j) pkv[j] = f2bfh(acc[j]);
    *(v8bf*)(Wp + bidx) = pkv;     // one contiguous 16B LDS store
  }
  if (tid < 128) {   // b' = qb @ blockdiag(M)
    int h = tid >> 4, jj = tid & 15;
    float s = 0.f;
    for (int i = 0; i < 16; ++i)
      s += qbias[h * 16 + i] * bf2f(Ms[h * 256 + i * 16 + jj]);
    bprime[tid] = s;
  }
  __syncthreads();

  // ---- Phase C: 576 query tokens in 36 row tiles; wave w = column tile w ----
  for (int rt2 = 0; rt2 < 36; ++rt2) {
    for (int idx = tid; idx < 16 * 128; idx += 256) {
      int t = idx & 15, ch = idx >> 4;
      int tok = rt2 * 16 + t;
      int hh = wy * 24 + tok / 24, ww = wx * 24 + tok % 24;
      float v = x2[((bi * 128 + ch) * 192 + hh) * 192 + ww] + posq[tok * 128 + ch];
      Ast[t * 128 + ch] = f2bf(v);
    }
    __syncthreads();
    v8f acc = {};
    for (int k0 = 0; k0 < 128; k0 += 32)
      acc = wmma_step(frag_a_v(Ast, 128, k0, lane),
                      frag_b_pk(Wp, wave, k0 >> 5, 4, lane), acc);
    int n = lane & 15, mb = (lane < 16) ? 0 : 8;
#pragma unroll
    for (int j = 0; j < 8; ++j) {
      int tok = rt2 * 16 + mb + j;
      int col = wave * 16 + n;
      int hh = wy * 24 + tok / 24, ww = wx * 24 + tok % 24;
      float res = x2[((bi * 128 + col) * 192 + hh) * 192 + ww] + acc[j] + bprime[col];
      qbuf[((bi * 192 + hh) * 192 + ww) * 128 + col] = res;
    }
    __syncthreads();
  }
}

// ---------------------------------------------------------------------------
// x = x + LN(x) @ W + b   (proj1 / proj2 residual), rows = B*192*192, K=N=128
// ---------------------------------------------------------------------------
__global__ __launch_bounds__(256) void k_ln_gemm_res(
    float* __restrict__ xio, const unsigned short* __restrict__ W_pk,
    const float* __restrict__ bias, const float* __restrict__ gamma,
    const float* __restrict__ beta) {
  __shared__ __align__(16) char smem[24576];
  float* xS = (float*)smem;                              // 32 x 128 f32
  unsigned short* aS = (unsigned short*)(smem + 16384);  // 32 x 128 bf16 (normalized)
  int tid = threadIdx.x, lane = tid & 31, wave = tid >> 5;
  int rows0 = blockIdx.x * 32;
  for (int idx = tid; idx < 32 * 128; idx += 256)
    xS[idx] = xio[(rows0 + (idx >> 7)) * 128 + (idx & 127)];
  __syncthreads();
  if (tid < 32) {
    float mu = 0.f, var = 0.f;
    for (int c = 0; c < 128; ++c) mu += xS[tid * 128 + c];
    mu *= (1.f / 128.f);
    for (int c = 0; c < 128; ++c) { float d = xS[tid * 128 + c] - mu; var += d * d; }
    var *= (1.f / 128.f);
    float inv = rsqrtf(var + 1e-5f);
    for (int c = 0; c < 128; ++c)
      aS[tid * 128 + c] = f2bf((xS[tid * 128 + c] - mu) * inv * gamma[c] + beta[c]);
  }
  __syncthreads();
  for (int job = wave; job < 16; job += 8) {
    int rt = job >> 3, ct = job & 7;
    v8f acc = {};
    for (int k0 = 0; k0 < 128; k0 += 32)
      acc = wmma_step(frag_a_v(aS + rt * 16 * 128, 128, k0, lane),
                      frag_b_pk(W_pk, ct, k0 >> 5, 4, lane), acc);
    int n = lane & 15, mb = (lane < 16) ? 0 : 8;
#pragma unroll
    for (int j = 0; j < 8; ++j) {
      int row = rt * 16 + mb + j, col = ct * 16 + n;
      xio[(rows0 + row) * 128 + col] = xS[row * 128 + col] + acc[j] + bias[col];
    }
  }
}

// ---------------------------------------------------------------------------
// Small helper kernels (global attention path is tiny: 64 tokens/batch).
// ---------------------------------------------------------------------------
__global__ void k_pool1(const float* __restrict__ x1, float* __restrict__ pooled) {
  int i = blockIdx.x * blockDim.x + threadIdx.x;
  if (i >= 2 * 64 * 256) return;
  int bi = i / (64 * 256), tok = (i >> 8) & 63, ch = i & 255;
  int gy = tok >> 3, gx = tok & 7;
  float s = 0.f;
  for (int iy = 0; iy < 12; ++iy)
    for (int ix = 0; ix < 12; ++ix)
      s += x1[((bi * 256 + ch) * 96 + gy * 12 + iy) * 96 + gx * 12 + ix];
  pooled[i] = s * (1.f / 144.f);
}

__global__ void k_poolq(const float* __restrict__ qbuf, float* __restrict__ pooled) {
  int i = blockIdx.x * blockDim.x + threadIdx.x;
  if (i >= 2 * 64 * 128) return;
  int bi = i / (64 * 128), tok = (i >> 7) & 63, ch = i & 127;
  int gy = tok >> 3, gx = tok & 7;
  float s = 0.f;
  for (int iy = 0; iy < 24; ++iy)
    for (int ix = 0; ix < 24; ++ix)
      s += qbuf[((bi * 192 + gy * 24 + iy) * 192 + gx * 24 + ix) * 128 + ch];
  pooled[i] = s * (1.f / 576.f);
}

// out[t,d] = bias[d] + sum_c in[t,c] W[c,d] (+ posg[t%64, d])
__global__ void k_gemm_small(const float* __restrict__ in, const float* __restrict__ W,
                             const float* __restrict__ bias, const float* __restrict__ pos,
                             float* __restrict__ out, int K, int T) {
  int i = blockIdx.x * blockDim.x + threadIdx.x;
  if (i >= T * 128) return;
  int t = i >> 7, d = i & 127;
  float s = bias[d];
  for (int c = 0; c < K; ++c) s += in[t * K + c] * W[c * 128 + d];
  if (pos) s += pos[(t & 63) * 128 + d];
  out[i] = s;
}

__global__ void k_global_attn(const float* __restrict__ q2, const float* __restrict__ k2,
                              const float* __restrict__ v2, float* __restrict__ og) {
  int bi = blockIdx.x >> 3, h = blockIdx.x & 7;
  int n = threadIdx.x;  // query token 0..63
  float qv[16];
  for (int j = 0; j < 16; ++j) qv[j] = q2[(bi * 64 + n) * 128 + h * 16 + j];
  // pass 1: max
  float mx = -1e30f;
  for (int m = 0; m < 64; ++m) {
    float d = 0.f;
    for (int j = 0; j < 16; ++j) d += qv[j] * k2[(bi * 64 + m) * 128 + h * 16 + j];
    mx = fmaxf(mx, d);
  }
  // pass 2: exp-sum + weighted V accumulation (no score array -> no spill)
  float o[16];
  for (int j = 0; j < 16; ++j) o[j] = 0.f;
  float sum = 0.f;
  for (int m = 0; m < 64; ++m) {
    float d = 0.f;
    for (int j = 0; j < 16; ++j) d += qv[j] * k2[(bi * 64 + m) * 128 + h * 16 + j];
    float e = expf(d - mx);
    sum += e;
    for (int j = 0; j < 16; ++j) o[j] += e * v2[(bi * 64 + m) * 128 + h * 16 + j];
  }
  float inv = 1.f / sum;
  for (int j = 0; j < 16; ++j)
    og[(bi * 64 + n) * 128 + h * 16 + j] = o[j] * inv;
}

__global__ void k_add_og(float* __restrict__ qbuf, const float* __restrict__ og) {
  int i = blockIdx.x * blockDim.x + threadIdx.x;
  if (i >= 2 * 192 * 192 * 128) return;
  int bi = i / (192 * 192 * 128);
  int rem = i % (192 * 192 * 128);
  int hh = rem / (192 * 128), ww = (rem / 128) % 192, c = rem & 127;
  float fy = hh * (7.f / 191.f); int y0 = (int)fy; int y1 = min(y0 + 1, 7); float ty = fy - y0;
  float fx = ww * (7.f / 191.f); int x0 = (int)fx; int x1i = min(x0 + 1, 7); float tx = fx - x0;
#define OGV(Y, X) og[(bi * 64 + (Y) * 8 + (X)) * 128 + c]
  float v = (1.f - ty) * ((1.f - tx) * OGV(y0, x0) + tx * OGV(y0, x1i)) +
            ty * ((1.f - tx) * OGV(y1, x0) + tx * OGV(y1, x1i));
#undef OGV
  qbuf[i] += v;
}

__global__ void k_final(const float* __restrict__ qbuf, const float* __restrict__ shortcut,
                        float* __restrict__ out) {
  int i = blockIdx.x * blockDim.x + threadIdx.x;
  if (i >= 2 * 128 * 192 * 192) return;
  int bi = i / (128 * 192 * 192);
  int rem = i % (128 * 192 * 192);
  int c = rem / (192 * 192);
  int hh = (rem / 192) % 192, ww = rem % 192;
  float fy = hh * (95.f / 191.f); int y0 = (int)fy; int y1 = min(y0 + 1, 95); float ty = fy - y0;
  float fx = ww * (95.f / 191.f); int x0 = (int)fx; int x1i = min(x0 + 1, 95); float tx = fx - x0;
#define SCV(Y, X) shortcut[((bi * 96 + (Y)) * 96 + (X)) * 128 + c]
  float sc = (1.f - ty) * ((1.f - tx) * SCV(y0, x0) + tx * SCV(y0, x1i)) +
             ty * ((1.f - tx) * SCV(y1, x0) + tx * SCV(y1, x1i));
#undef SCV
  out[i] = qbuf[((bi * 192 + hh) * 192 + ww) * 128 + c] + sc;
}

// ---------------------------------------------------------------------------
extern "C" void kernel_launch(void* const* d_in, const int* in_sizes, int n_in,
                              void* d_out, int out_size, void* d_ws, size_t ws_size,
                              hipStream_t stream) {
  const float* x1      = (const float*)d_in[0];
  const float* x2      = (const float*)d_in[1];
  const float* proj_w  = (const float*)d_in[2];
  const float* proj_b  = (const float*)d_in[3];
  const float* proj_g  = (const float*)d_in[4];
  const float* proj_bt = (const float*)d_in[5];
  const float* qw      = (const float*)d_in[6];
  const float* qb      = (const float*)d_in[7];
  const float* kw      = (const float*)d_in[8];
  const float* kb      = (const float*)d_in[9];
  const float* vw      = (const float*)d_in[10];
  const float* vb      = (const float*)d_in[11];
  const float* p1_g    = (const float*)d_in[12];
  const float* p1_bt   = (const float*)d_in[13];
  const float* p1_w    = (const float*)d_in[14];
  const float* p1_b    = (const float*)d_in[15];
  const float* apq_w   = (const float*)d_in[16];
  const float* apq_b   = (const float*)d_in[17];
  const float* apk_w   = (const float*)d_in[18];
  const float* apk_b   = (const float*)d_in[19];
  const float* q2w     = (const float*)d_in[20];
  const float* q2b     = (const float*)d_in[21];
  const float* k2w     = (const float*)d_in[22];
  const float* k2b     = (const float*)d_in[23];
  const float* v2w     = (const float*)d_in[24];
  const float* v2b     = (const float*)d_in[25];
  const float* p2_g    = (const float*)d_in[26];
  const float* p2_bt   = (const float*)d_in[27];
  const float* p2_w    = (const float*)d_in[28];
  const float* p2_b    = (const float*)d_in[29];

  char* ws = (char*)d_ws;
  float* qbuf      = (float*)(ws + 0);          // 2*192*192*128 f32
  float* shortcut  = (float*)(ws + 37748736);   // 2*96*96*128 f32
  float* posq      = (float*)(ws + 47185920);
  float* posk      = (float*)(ws + 47480832);
  float* posg      = (float*)(ws + 47628288);
  float* pooled1   = (float*)(ws + 47661056);
  float* kg        = (float*)(ws + 47792128);
  float* k2b_      = (float*)(ws + 47857664);
  float* v2b_      = (float*)(ws + 47923200);
  float* pooledq   = (float*)(ws + 47988736);
  float* qg        = (float*)(ws + 48054272);
  float* q2b_      = (float*)(ws + 48119808);
  float* og        = (float*)(ws + 48185344);
  unsigned short* projw_pk = (unsigned short*)(ws + 48250880);
  unsigned short* kw_pk    = (unsigned short*)(ws + 48316416);
  unsigned short* vw_pk    = (unsigned short*)(ws + 48381952);
  unsigned short* qw_bf    = (unsigned short*)(ws + 48447488);
  unsigned short* p1w_pk   = (unsigned short*)(ws + 48480256);
  unsigned short* p2w_pk   = (unsigned short*)(ws + 48513024);
  float* out = (float*)d_out;

  k_pos_init<<<(73728 + 36864 + 8192 + 255) / 256, 256, 0, stream>>>(posq, posk, posg);
  k_pack_b<<<128, 256, 0, stream>>>(proj_w, projw_pk, 256);
  k_pack_b<<<128, 256, 0, stream>>>(kw, kw_pk, 256);
  k_pack_b<<<128, 256, 0, stream>>>(vw, vw_pk, 256);
  k_pack_b<<<64, 256, 0, stream>>>(p1_w, p1w_pk, 128);
  k_pack_b<<<64, 256, 0, stream>>>(p2_w, p2w_pk, 128);
  k_cvt_bf16<<<64, 256, 0, stream>>>(qw, qw_bf, 16384);   // A-side: plain row-major

  // shortcut branch (WMMA + LN)
  k_proj_ln<<<576, 256, 0, stream>>>(x1, projw_pk, proj_b, proj_g, proj_bt, shortcut);

  // global-attention K/V (tiny)
  k_pool1<<<128, 256, 0, stream>>>(x1, pooled1);
  k_gemm_small<<<64, 256, 0, stream>>>(pooled1, apk_w, apk_b, posg, kg, 256, 128);
  k_gemm_small<<<64, 256, 0, stream>>>(kg, k2w, k2b, nullptr, k2b_, 128, 128);
  k_gemm_small<<<64, 256, 0, stream>>>(kg, v2w, v2b, nullptr, v2b_, 128, 128);

  // local windowed linear attention (WMMA), then proj1 residual (WMMA)
  k_local_attn<<<128, 256, 0, stream>>>(x1, x2, kw_pk, kb, vw_pk, vb, qw_bf, qb,
                                        posk, posq, qbuf);
  k_ln_gemm_res<<<2304, 256, 0, stream>>>(qbuf, p1w_pk, p1_b, p1_g, p1_bt);

  // global attention over pooled tokens
  k_poolq<<<64, 256, 0, stream>>>(qbuf, pooledq);
  k_gemm_small<<<64, 256, 0, stream>>>(pooledq, apq_w, apq_b, posg, qg, 128, 128);
  k_gemm_small<<<64, 256, 0, stream>>>(qg, q2w, q2b, nullptr, q2b_, 128, 128);
  k_global_attn<<<16, 64, 0, stream>>>(q2b_, k2b_, v2b_, og);
  k_add_og<<<36864, 256, 0, stream>>>(qbuf, og);

  // proj2 residual (WMMA), then final add + layout transpose
  k_ln_gemm_res<<<2304, 256, 0, stream>>>(qbuf, p2w_pk, p2_b, p2_g, p2_bt);
  k_final<<<36864, 256, 0, stream>>>(qbuf, shortcut, out);

  (void)in_sizes; (void)n_in; (void)out_size; (void)ws_size;
}
